// SelfAttention_62534723830169
// MI455X (gfx1250) — matrix operand order repaired and checked by reading, not compiled, not run on hardware
//
#include <hip/hip_runtime.h>
#include <hip/hip_bf16.h>

typedef __attribute__((ext_vector_type(16))) __bf16 v16bf;
typedef __attribute__((ext_vector_type(8)))  __bf16 v8bf;
typedef __attribute__((ext_vector_type(8)))  float  v8f;

#define H_DIM   1024
#define NHEADS  16
#define HEAD_DIM 64
#define SEQ     1024
#define BATCH   8
#define ROWS    (BATCH * SEQ)   // 8192

// ---------------------------------------------------------------------------
// WMMA helpers (wave32, 16x16x32 bf16 -> f32)
// ---------------------------------------------------------------------------
__device__ __forceinline__ v8f wmma_bf16(v16bf a, v16bf b, v8f c) {
  return __builtin_amdgcn_wmma_f32_16x16x32_bf16(false, a, false, b,
                                                 (short)0, c, false, false);
}

// A fragment: 16(M) x 32(K), row-major source. lane: m = lane&15, h = lane>>4
// element e -> k = 8h + (e<8 ? e : e+8)  => two contiguous 16B chunks.
__device__ __forceinline__ v16bf load_a_frag(const __bf16* __restrict__ p,
                                             int ld, int row0, int k0,
                                             int m, int h) {
  const __bf16* r = p + (size_t)(row0 + m) * ld + k0 + h * 8;
  v8bf lo = *(const v8bf*)(r);
  v8bf hi = *(const v8bf*)(r + 16);
  v16bf o;
#pragma unroll
  for (int i = 0; i < 8; ++i) { o[i] = lo[i]; o[i + 8] = hi[i]; }
  return o;
}

// B fragment: 32(K) x 16(N), where memory is row-major with row = N index
// (i.e. B[k][n] = p[(n0+n)*ld + k0 + k]). lane: n = lane&15, h = lane>>4,
// element e -> k = 8h + (e<8 ? e : e+8)  => two contiguous 16B chunks.
__device__ __forceinline__ v16bf load_b_frag(const __bf16* __restrict__ p,
                                             int ld, int n0, int k0,
                                             int n, int h) {
  const __bf16* r = p + (size_t)(n0 + n) * ld + k0 + h * 8;
  v8bf lo = *(const v8bf*)(r);
  v8bf hi = *(const v8bf*)(r + 16);
  v16bf o;
#pragma unroll
  for (int i = 0; i < 8; ++i) { o[i] = lo[i]; o[i + 8] = hi[i]; }
  return o;
}

// ---------------------------------------------------------------------------
// Kernel 1: x = LayerNorm(hidden + pos_emb)  -> bf16 [ROWS, H]
// ---------------------------------------------------------------------------
__global__ __launch_bounds__(256) void ln_posadd_kernel(
    const float* __restrict__ hs, const float* __restrict__ pe,
    const float* __restrict__ lw, const float* __restrict__ lb,
    __bf16* __restrict__ xbf) {
  int row = blockIdx.x;
  int j = threadIdx.x;
  size_t base = (size_t)row * H_DIM;
  float4 a = ((const float4*)(hs + base))[j];
  float4 p = ((const float4*)(pe + (size_t)(row & (SEQ - 1)) * H_DIM))[j];
  float x0 = a.x + p.x, x1 = a.y + p.y, x2 = a.z + p.z, x3 = a.w + p.w;
  float s = x0 + x1 + x2 + x3;
  float q = x0 * x0 + x1 * x1 + x2 * x2 + x3 * x3;
#pragma unroll
  for (int off = 16; off > 0; off >>= 1) {
    s += __shfl_xor(s, off, 32);
    q += __shfl_xor(q, off, 32);
  }
  __shared__ float ssum[8], ssq[8];
  int wid = j >> 5, lane = j & 31;
  if (lane == 0) { ssum[wid] = s; ssq[wid] = q; }
  __syncthreads();
  float ts = 0.f, tq = 0.f;
#pragma unroll
  for (int i = 0; i < 8; ++i) { ts += ssum[i]; tq += ssq[i]; }
  float mu = ts * (1.0f / H_DIM);
  float var = tq * (1.0f / H_DIM) - mu * mu;
  float rstd = rsqrtf(var + 1e-12f);
  int c = j * 4;
  __bf16* o = xbf + base + c;
  o[0] = (__bf16)((x0 - mu) * rstd * lw[c + 0] + lb[c + 0]);
  o[1] = (__bf16)((x1 - mu) * rstd * lw[c + 1] + lb[c + 1]);
  o[2] = (__bf16)((x2 - mu) * rstd * lw[c + 2] + lb[c + 2]);
  o[3] = (__bf16)((x3 - mu) * rstd * lw[c + 3] + lb[c + 3]);
}

// ---------------------------------------------------------------------------
// Kernel 2: fp32 -> bf16 weight conversion (wq|wk|wv packed)
// ---------------------------------------------------------------------------
__global__ __launch_bounds__(256) void cvt_weights(
    const float* __restrict__ wq, const float* __restrict__ wk,
    const float* __restrict__ wv, __bf16* __restrict__ out) {
  size_t i = (size_t)blockIdx.x * blockDim.x + threadIdx.x;
  const size_t N = (size_t)H_DIM * H_DIM;
  out[i]         = (__bf16)wq[i];
  out[i + N]     = (__bf16)wk[i];
  out[i + 2 * N] = (__bf16)wv[i];
}

// ---------------------------------------------------------------------------
// Kernel 3: Y = X @ W^T + bias, head-split output.
// WG = 8 waves, 128x128 output block. Per 32-wide k-step the WG stages the
// 128x32 X tile and 128x32 W tile into LDS with GLOBAL_LOAD_ASYNC_TO_LDS_B128
// (ASYNCcnt), double-buffered. The double buffering is manually unrolled
// (phase 0 / phase 1 with compile-time buffer indices) so LDS offsets are
// immediates and accumulators stay pinned (no parity-merge register copies).
//   transposeOut==0 : Y[b][head][s][d]   (Q, K)
//   transposeOut==1 : Y[b][head][d][s]   (V -> V^T, K-contiguous for ctx GEMM)
// ---------------------------------------------------------------------------
__global__ __launch_bounds__(256) void qkv_gemm(
    const __bf16* __restrict__ X,    // [ROWS, H]
    const __bf16* __restrict__ W,    // [H, H] row-major, row = output column
    const float*  __restrict__ bias, // [H]
    __bf16* __restrict__ Y, int transposeOut) {
  // smem[buf][0] = X tile 128x32, smem[buf][1] = W tile 128x32  (32 KiB total)
  __shared__ __bf16 smem[2][2][128 * 32];

  const int tid = threadIdx.x;
  const int lane = tid & 31, wave = tid >> 5;
  const int ln = lane & 15, h = lane >> 4;
  const int wm = wave >> 1, wn = wave & 1;
  const int rowBase = blockIdx.x * 128;
  const int colBase = blockIdx.y * 128;

  const __bf16* Xblk = X + (size_t)rowBase * H_DIM;
  const __bf16* Wblk = W + (size_t)colBase * H_DIM;
  const unsigned ldsBase = (unsigned)(uintptr_t)(&smem[0][0][0]);

  // Each thread DMAs 2x16B per tile (512 chunks of 16B cover 128x32 bf16).
  auto issue_tiles = [&](int buf, int k) {
    const unsigned xOff = ldsBase + (unsigned)buf * 16384u;
    const unsigned wOff = xOff + 8192u;
#pragma unroll
    for (int c = 0; c < 2; ++c) {
      int lin = tid + c * 256;            // chunk id, 16B each
      int row = lin >> 2;                 // 4 chunks per 64B row
      int col = (lin & 3) * 8;            // element offset within row
      const __bf16* gx = Xblk + (size_t)row * H_DIM + k + col;
      const __bf16* gw = Wblk + (size_t)row * H_DIM + k + col;
      unsigned lx = xOff + (unsigned)lin * 16u;
      unsigned lw2 = wOff + (unsigned)lin * 16u;
      asm volatile("global_load_async_to_lds_b128 %0, %1, off"
                   :: "v"(lx), "v"(gx) : "memory");
      asm volatile("global_load_async_to_lds_b128 %0, %1, off"
                   :: "v"(lw2), "v"(gw) : "memory");
    }
  };

  v8f acc[2][4];
#pragma unroll
  for (int i = 0; i < 2; ++i)
#pragma unroll
    for (int t = 0; t < 4; ++t)
#pragma unroll
      for (int r = 0; r < 8; ++r) acc[i][t][r] = 0.0f;

  // Compute one 32-wide k-step from buffer `buf` (compile-time constant).
  auto compute = [&](int buf) {
    const __bf16* xt = &smem[buf][0][0];
    const __bf16* wt = &smem[buf][1][0];
    v16bf a0 = load_a_frag(xt, 32, wm * 32,      0, ln, h);
    v16bf a1 = load_a_frag(xt, 32, wm * 32 + 16, 0, ln, h);
#pragma unroll
    for (int t = 0; t < 4; ++t) {
      v16bf bt = load_b_frag(wt, 32, wn * 64 + t * 16, 0, ln, h);
      acc[0][t] = wmma_bf16(a0, bt, acc[0][t]);
      acc[1][t] = wmma_bf16(a1, bt, acc[1][t]);
    }
  };

  issue_tiles(0, 0);

#pragma unroll 1
  for (int k = 0; k < H_DIM; k += 64) {
    // ---- phase 0: consume buf 0, prefetch buf 1 (k+32 <= 992, always valid)
    asm volatile("s_wait_asynccnt 0x0" ::: "memory");  // own DMA chunks landed
    __syncthreads();                                   // whole tile landed
    issue_tiles(1, k + 32);
    compute(0);
    // ---- phase 1: consume buf 1, prefetch buf 0
    asm volatile("s_wait_asynccnt 0x0" ::: "memory");
    __syncthreads();
    if (k + 64 < H_DIM) issue_tiles(0, k + 64);
    compute(1);
  }

#pragma unroll
  for (int t = 0; t < 4; ++t) {
    int col = colBase + wn * 64 + t * 16 + ln;
    float bv = bias[col];
    int head = col >> 6, d = col & (HEAD_DIM - 1);
#pragma unroll
    for (int i = 0; i < 2; ++i) {
#pragma unroll
      for (int r = 0; r < 8; ++r) {
        int row = rowBase + wm * 32 + i * 16 + r + 8 * h;  // C/D: M = r + 8h
        int b = row >> 10, s = row & (SEQ - 1);
        float v = acc[i][t][r] + bv;
        size_t idx;
        if (transposeOut)
          idx = (((size_t)(b * NHEADS + head)) * HEAD_DIM + d) * SEQ + s;
        else
          idx = (((size_t)(b * NHEADS + head)) * SEQ + s) * HEAD_DIM + d;
        Y[idx] = (__bf16)v;
      }
    }
  }
}

// ---------------------------------------------------------------------------
// Kernel 4: flash attention. 4 waves/WG, each wave owns 16 query rows.
// Per 32-key block: 2 score WMMAs (K=32 over d, x2 for 64) + online softmax
// + P staged via LDS into A-fragment layout + 4 ctx WMMAs against V^T.
// ---------------------------------------------------------------------------
__global__ __launch_bounds__(128) void attn_kernel(
    const __bf16* __restrict__ Q,   // [B, NH, S, HD]
    const __bf16* __restrict__ K,   // [B, NH, S, HD]
    const __bf16* __restrict__ Vt,  // [B, NH, HD, S]
    float* __restrict__ out) {      // [B, S, H]
  const int tid = threadIdx.x;
  const int lane = tid & 31, wave = tid >> 5;
  const int ln = lane & 15, h = lane >> 4;
  const int qb = blockIdx.x, head = blockIdx.y, b = blockIdx.z;
  const int qBase = qb * 64 + wave * 16;

  const size_t hoff = ((size_t)b * NHEADS + head) * SEQ * HEAD_DIM;
  const __bf16* Qh = Q + hoff;
  const __bf16* Kh = K + hoff;
  const __bf16* Vh = Vt + hoff;    // [HD, S]

  __shared__ __bf16 pst[4][16 * 32];

  v16bf qf0 = load_a_frag(Qh, HEAD_DIM, qBase, 0,  ln, h);
  v16bf qf1 = load_a_frag(Qh, HEAD_DIM, qBase, 32, ln, h);

  float rowm[8], rowl[8];
  v8f o[4];
#pragma unroll
  for (int r = 0; r < 8; ++r) { rowm[r] = -1e30f; rowl[r] = 0.0f; }
#pragma unroll
  for (int t = 0; t < 4; ++t)
#pragma unroll
    for (int r = 0; r < 8; ++r) o[t][r] = 0.0f;

  for (int kb = 0; kb < SEQ; kb += 32) {
    v8f s0, s1;
#pragma unroll
    for (int r = 0; r < 8; ++r) { s0[r] = 0.0f; s1[r] = 0.0f; }
    s0 = wmma_bf16(qf0, load_b_frag(Kh, HEAD_DIM, kb,      0,  ln, h), s0);
    s0 = wmma_bf16(qf1, load_b_frag(Kh, HEAD_DIM, kb,      32, ln, h), s0);
    s1 = wmma_bf16(qf0, load_b_frag(Kh, HEAD_DIM, kb + 16, 0,  ln, h), s1);
    s1 = wmma_bf16(qf1, load_b_frag(Kh, HEAD_DIM, kb + 16, 32, ln, h), s1);

    float alpha[8];
#pragma unroll
    for (int r = 0; r < 8; ++r) {
      float a  = s0[r] * 0.125f;          // 1/sqrt(64)
      float bb = s1[r] * 0.125f;
      float mx = fmaxf(a, bb);
#pragma unroll
      for (int off = 8; off > 0; off >>= 1)      // stays inside 16-lane half
        mx = fmaxf(mx, __shfl_xor(mx, off, 32));
      float mnew = fmaxf(rowm[r], mx);
      alpha[r] = __expf(rowm[r] - mnew);
      rowm[r] = mnew;
      float p0 = __expf(a - mnew), p1 = __expf(bb - mnew);
      float rs = p0 + p1;
#pragma unroll
      for (int off = 8; off > 0; off >>= 1)
        rs += __shfl_xor(rs, off, 32);
      rowl[r] = rowl[r] * alpha[r] + rs;
      int prow = r + 8 * h;                      // C-layout row
      pst[wave][prow * 32 + ln]      = (__bf16)p0;
      pst[wave][prow * 32 + 16 + ln] = (__bf16)p1;
    }
    asm volatile("s_wait_dscnt 0" ::: "memory");

    // Re-read the 16x32 P tile from LDS in A-fragment layout.
    const __bf16* pr = &pst[wave][ln * 32 + h * 8];
    v8bf plo = *(const v8bf*)(pr);
    v8bf phi = *(const v8bf*)(pr + 16);
    v16bf pa;
#pragma unroll
    for (int i = 0; i < 8; ++i) { pa[i] = plo[i]; pa[i + 8] = phi[i]; }

#pragma unroll
    for (int t = 0; t < 4; ++t) {
#pragma unroll
      for (int r = 0; r < 8; ++r) o[t][r] *= alpha[r];
      o[t] = wmma_bf16(pa, load_b_frag(Vh, SEQ, t * 16, kb, ln, h), o[t]);
    }
  }

#pragma unroll
  for (int r = 0; r < 8; ++r) {
    float inv = 1.0f / rowl[r];
    int srow = qBase + r + 8 * h;
    float* orow = out + ((size_t)b * SEQ + srow) * H_DIM + head * HEAD_DIM;
#pragma unroll
    for (int t = 0; t < 4; ++t) orow[t * 16 + ln] = o[t][r] * inv;
  }
}

// ---------------------------------------------------------------------------
// Launch. Workspace layout (bytes):
//   [0,        16MiB)  xbf   bf16 [8192,1024]
//   [16MiB,    22MiB)  wbf   bf16 wq|wk|wv
//   [22MiB,    38MiB)  qbf   bf16 [B,NH,S,HD]
//   [38MiB,    54MiB)  kbf   bf16 [B,NH,S,HD]
//   [54MiB,    70MiB)  vtbf  bf16 [B,NH,HD,S]
// ---------------------------------------------------------------------------
extern "C" void kernel_launch(void* const* d_in, const int* in_sizes, int n_in,
                              void* d_out, int out_size, void* d_ws, size_t ws_size,
                              hipStream_t stream) {
  (void)in_sizes; (void)n_in; (void)out_size; (void)ws_size;
  const float* hs  = (const float*)d_in[0];
  const float* pe  = (const float*)d_in[1];
  const float* lnw = (const float*)d_in[2];
  const float* lnb = (const float*)d_in[3];
  const float* wq  = (const float*)d_in[4];
  const float* bq  = (const float*)d_in[5];
  const float* wk  = (const float*)d_in[6];
  const float* bk  = (const float*)d_in[7];
  const float* wv  = (const float*)d_in[8];
  const float* bv  = (const float*)d_in[9];
  float* out = (float*)d_out;

  char* ws = (char*)d_ws;
  __bf16* xbf  = (__bf16*)(ws);
  __bf16* wbf  = (__bf16*)(ws + 16777216);
  __bf16* qbf  = (__bf16*)(ws + 23068672);
  __bf16* kbf  = (__bf16*)(ws + 39845888);
  __bf16* vtbf = (__bf16*)(ws + 56623104);
  const size_t NW = (size_t)H_DIM * H_DIM;

  ln_posadd_kernel<<<ROWS, 256, 0, stream>>>(hs, pe, lnw, lnb, xbf);
  cvt_weights<<<4096, 256, 0, stream>>>(wq, wk, wv, wbf);
  qkv_gemm<<<dim3(64, 8), 256, 0, stream>>>(xbf, wbf,          bq, qbf, 0);
  qkv_gemm<<<dim3(64, 8), 256, 0, stream>>>(xbf, wbf + NW,     bk, kbf, 0);
  qkv_gemm<<<dim3(64, 8), 256, 0, stream>>>(xbf, wbf + 2 * NW, bv, vtbf, 1);
  attn_kernel<<<dim3(16, 16, 8), 128, 0, stream>>>(qbf, kbf, vtbf, out);
}
